// MGFSoftmax_61289183314191
// MI455X (gfx1250) — matrix operand order repaired and checked by reading, not compiled, not run on hardware
//
#include <hip/hip_runtime.h>
#include <hip/hip_bf16.h>

typedef __attribute__((ext_vector_type(16))) _Float16 v16h;
typedef __attribute__((ext_vector_type(8)))  _Float16 v8h;
typedef __attribute__((ext_vector_type(8)))  float    v8f;

#define N_SEQ 1024
#define DHEAD 64
#define QTILE 128           // Q rows per workgroup (8 waves x 16 rows)
#define KCHUNK 32           // kv rows per streamed chunk

// Degree-7 truncated-Taylor exp, Horner form (matches reference inv_factorials).
__device__ __forceinline__ float poly_exp7(float x) {
    float acc = 1.98412698412698412698e-4f;   // 1/5040
    acc = acc * x + 1.38888888888888888889e-3f; // 1/720
    acc = acc * x + 8.33333333333333333333e-3f; // 1/120
    acc = acc * x + 4.16666666666666666667e-2f; // 1/24
    acc = acc * x + 1.66666666666666666667e-1f; // 1/6
    acc = acc * x + 0.5f;
    acc = acc * x + 1.0f;
    acc = acc * x + 1.0f;
    return acc;
}

__global__ __launch_bounds__(256)
void polyexp_attn_kernel(const float* __restrict__ q,
                         const float* __restrict__ k,
                         const float* __restrict__ v,
                         float* __restrict__ out)
{
    // LDS staging: K chunk row-major f16, V chunk transposed (d-major) f16,
    // per-wave P buffer for accumulator->A-fragment reshape.
    __shared__ _Float16 klds[KCHUNK][DHEAD];      // 4 KB
    __shared__ _Float16 vldsT[DHEAD][KCHUNK];     // 4 KB
    __shared__ _Float16 plds[8][16][KCHUNK];      // 8 KB

    const int bh    = blockIdx.y;                 // 0..B*H-1
    const int qbase = blockIdx.x * QTILE;
    const int tid   = threadIdx.x;
    const int wave  = tid >> 5;
    const int lane  = tid & 31;
    const int lm    = lane & 15;                  // col / row-in-tile index
    const int lh    = lane >> 4;                  // lane-half select

    const size_t base = (size_t)bh * (size_t)(N_SEQ * DHEAD);
    const float* qb = q + base;
    const float* kb = k + base;
    const float* vb = v + base;
    float*       ob = out + base;

    const int m0 = qbase + wave * 16;             // first Q row of this wave

    // ---- Q A-fragments (16x32 f16), K-dim split into t=0,1 -----------------
    // A layout (16-bit A 16x32, contiguous k order used by this kernel): lane holds row M=lane%16;
    // element e -> K = t*32 + lh*16 + e
    v16h qa[2];
    {
        const float* qrow = qb + (size_t)(m0 + lm) * DHEAD;
        #pragma unroll
        for (int t = 0; t < 2; ++t) {
            #pragma unroll
            for (int i = 0; i < 8; ++i) {
                qa[t][i]     = (_Float16)qrow[t * 32 + lh * 16 + i];
                qa[t][i + 8] = (_Float16)qrow[t * 32 + lh * 16 + 8 + i];
            }
        }
    }

    // All-ones B fragment: rowsum via WMMA (P x ones -> row sums replicated
    // across columns, same row mapping as the O accumulators).
    v16h ones;
    #pragma unroll
    for (int i = 0; i < 16; ++i) ones[i] = (_Float16)1.0f;

    v8f oacc0 = {}, oacc1 = {}, oacc2 = {}, oacc3 = {};
    v8f rsum  = {};

    for (int j0 = 0; j0 < N_SEQ; j0 += KCHUNK) {
        __syncthreads();   // previous-iteration readers done before restage

        // ---- Cooperative staging: 32x64 K and V chunks, fp32 -> f16 --------
        #pragma unroll
        for (int e = 0; e < 8; ++e) {
            int idx = e * 256 + tid;              // 0..2047, coalesced
            int row = idx >> 6;                   // 0..31
            int d   = idx & 63;                   // 0..63
            klds[row][d]  = (_Float16)kb[(size_t)(j0 + row) * DHEAD + d];
            vldsT[d][row] = (_Float16)vb[(size_t)(j0 + row) * DHEAD + d];
        }
        if (j0 + KCHUNK < N_SEQ) {
            // Pull next chunk toward L2 (global_prefetch_b8).
            __builtin_prefetch(&kb[(size_t)(j0 + KCHUNK) * DHEAD + tid * 8], 0, 1);
            __builtin_prefetch(&vb[(size_t)(j0 + KCHUNK) * DHEAD + tid * 8], 0, 1);
        }
        __syncthreads();

        // ---- S = Q . K^T for two 16-col tiles; poly-exp; spill P to LDS ----
        #pragma unroll
        for (int c = 0; c < 2; ++c) {
            v8f s = {};
            #pragma unroll
            for (int t = 0; t < 2; ++t) {
                // B layout (32x16 f16): lane col N=lane%16, element e -> K = lh*16 + e
                v16h bk = *(const v16h*)&klds[c * 16 + lm][t * 32 + lh * 16];
                s = __builtin_amdgcn_wmma_f32_16x16x32_f16(
                        false, qa[t], false, bk, (short)0, s, false, false);
            }
            // D layout: vgpr r, lane -> S[row = r + 8*lh][col = c*16 + lm]
            #pragma unroll
            for (int r = 0; r < 8; ++r) {
                float ex = poly_exp7(s[r]);
                plds[wave][r + lh * 8][c * 16 + lm] = (_Float16)ex;
            }
        }
        // Same-wave DS ops are in-order; compiler inserts s_wait_dscnt on the
        // store->load dependency through plds.

        // ---- Reload P as A-fragment (16x32 f16) ----------------------------
        v16h pa = *(const v16h*)&plds[wave][lm][lh * 16];

        // rowsum += P x ones
        rsum = __builtin_amdgcn_wmma_f32_16x16x32_f16(
                   false, pa, false, ones, (short)0, rsum, false, false);

        // ---- O += P x V, four 16-wide d-column groups ----------------------
        {
            v16h b0 = *(const v16h*)&vldsT[ 0 + lm][lh * 16];
            oacc0 = __builtin_amdgcn_wmma_f32_16x16x32_f16(
                        false, pa, false, b0, (short)0, oacc0, false, false);
            v16h b1 = *(const v16h*)&vldsT[16 + lm][lh * 16];
            oacc1 = __builtin_amdgcn_wmma_f32_16x16x32_f16(
                        false, pa, false, b1, (short)0, oacc1, false, false);
            v16h b2 = *(const v16h*)&vldsT[32 + lm][lh * 16];
            oacc2 = __builtin_amdgcn_wmma_f32_16x16x32_f16(
                        false, pa, false, b2, (short)0, oacc2, false, false);
            v16h b3 = *(const v16h*)&vldsT[48 + lm][lh * 16];
            oacc3 = __builtin_amdgcn_wmma_f32_16x16x32_f16(
                        false, pa, false, b3, (short)0, oacc3, false, false);
        }
    }

    // ---- Normalize and store: rsum shares the accumulator row mapping ------
    #pragma unroll
    for (int r = 0; r < 8; ++r) {
        const float inv = 1.0f / (rsum[r] + 1e-6f);
        const size_t rowoff = (size_t)(m0 + r + lh * 8) * DHEAD + lm;
        ob[rowoff +  0] = oacc0[r] * inv;
        ob[rowoff + 16] = oacc1[r] * inv;
        ob[rowoff + 32] = oacc2[r] * inv;
        ob[rowoff + 48] = oacc3[r] * inv;
    }
}

extern "C" void kernel_launch(void* const* d_in, const int* in_sizes, int n_in,
                              void* d_out, int out_size, void* d_ws, size_t ws_size,
                              hipStream_t stream) {
    (void)in_sizes; (void)n_in; (void)d_ws; (void)ws_size; (void)out_size;
    const float* q = (const float*)d_in[0];
    const float* k = (const float*)d_in[1];
    const float* v = (const float*)d_in[2];
    // d_in[3] = inv_factorials: fixed degree-7 table, folded into constants.
    float* out = (float*)d_out;

    const int B = 8, H = 12;
    dim3 grid(N_SEQ / QTILE, B * H);   // (8, 96)
    dim3 block(256);                   // 8 waves of 32
    polyexp_attn_kernel<<<grid, block, 0, stream>>>(q, k, v, out);
}
